// Vectorizer_69337952026900
// MI455X (gfx1250) — compile-verified, matrix operand order checked by reading, not run on hardware
//
#include <hip/hip_runtime.h>

typedef __attribute__((ext_vector_type(16))) _Float16 v16h;
typedef __attribute__((ext_vector_type(8)))  float    v8f;
typedef __attribute__((ext_vector_type(4)))  unsigned int u32x4;
typedef __attribute__((ext_vector_type(8)))  int      i32x8;
typedef __attribute__((ext_vector_type(4)))  int      i32x4;

#define HID 768
#define NH 12
#define HD 64
#define FF 3072
#define WIN 256
#define PDIM 128

#if __has_builtin(__builtin_amdgcn_tensor_load_to_lds)
#define USE_TDM 1
#else
#define USE_TDM 0
#endif

// ---------------------------------------------------------------------------
// TDM: DMA a 2D f16 tile (32 elems x 128 rows) from global into LDS with a
// 1-dword pad after each 16-dword row => padded LDS row stride of 34 halves.
// Descriptor per CDNA5 ISA ch.8 (D# groups 0/1). Tensor dims are given
// relative to the tile start so rows/cols past the tensor read as zeros.
// ---------------------------------------------------------------------------
#if USE_TDM
__device__ __forceinline__ void tdm_load_tile_f16(
    unsigned lds_byte_off, const _Float16* gptr,
    int dim0_rem, int rows_rem, int row_stride_elems)
{
    unsigned long long ga = (unsigned long long)(size_t)gptr;
    u32x4 g0;
    g0[0] = 1u;                                   // count=1, user mode, no gather
    g0[1] = lds_byte_off;                         // LDS byte address
    g0[2] = (unsigned)ga;                         // global addr [31:0]
    g0[3] = (unsigned)((ga >> 32) & 0x01FFFFFFull) | (2u << 30);  // [56:32], type=2
    unsigned td0 = (unsigned)dim0_rem;
    unsigned td1 = (unsigned)(rows_rem < 0 ? 0 : rows_rem);
    i32x8 g1;
    g1[0] = (int)((1u << 16)    // data_size = 2 bytes
                | (1u << 20)    // pad_enable
                | (3u << 22)    // pad_interval = 16 dwords (one 32-half row)
                | (0u << 25));  // pad_amount  = 1 dword
    g1[1] = (int)((td0 & 0xFFFFu) << 16);                     // tensor_dim0 lo
    g1[2] = (int)((td0 >> 16) | ((td1 & 0xFFFFu) << 16));     // dim0 hi | dim1 lo
    g1[3] = (int)((td1 >> 16) | (32u << 16));                 // dim1 hi | tile_dim0=32
    g1[4] = 128;                                              // tile_dim1=128, tile_dim2=0
    g1[5] = row_stride_elems;                                 // tensor_dim0_stride lo
    g1[6] = 0;
    g1[7] = 0;
    i32x4 z4 = {0, 0, 0, 0};
#if defined(__clang_major__) && (__clang_major__ >= 23)
    i32x8 z8 = {0, 0, 0, 0, 0, 0, 0, 0};
    __builtin_amdgcn_tensor_load_to_lds(g0, g1, z4, z4, z8, 0);
#else
    __builtin_amdgcn_tensor_load_to_lds(g0, g1, z4, z4, 0);
#endif
}
#endif

// ---------------------------------------------------------------------------
// GEMM: C[M,N] = act(A[M,K] @ Wt[N,K]^T + bias). A, Wt are f16; accumulate
// fp32 via v_wmma_f32_16x16x32_f16. Block tile 128x128, 8 waves, each wave
// 32x64 (2x4 WMMA tiles). Tiles staged to LDS by the Tensor Data Mover,
// double-buffered, wave 0 issues DMA and synchronizes with s_wait_tensorcnt.
// act: 0=none, 1=exact GELU, 2=ReLU.  Outputs: Cf (fp32) and/or Ch (f16).
// LDS layout (bytes): A0@0, A1@8704, B0@17408, B1@26112 (34-half row stride).
// ---------------------------------------------------------------------------
#define LPAD 34

__global__ __launch_bounds__(256) void gemm_tdm_wmma(
    const _Float16* __restrict__ A, const _Float16* __restrict__ Wt,
    const float* __restrict__ bias, float* Cf, _Float16* Ch,
    int M, int N, int K, int act)
{
    extern __shared__ _Float16 lds[];
    const int row0 = blockIdx.y * 128;
    const int col0 = blockIdx.x * 128;
    const int tid = threadIdx.x;
    const int wave = tid >> 5;
    const int lane = tid & 31;
    const int lane_lo = lane & 15;
    const int hi = lane >> 4;
    const int wr = wave >> 1;   // 0..3 : 32-row group
    const int wc = wave & 1;    // 0..1 : 64-col group

    v8f acc[2][4] = {};
    const int nt = K / 32;

#if USE_TDM
    const unsigned lbase = (unsigned)(size_t)lds;
    if (wave == 0) {
        tdm_load_tile_f16(lbase,          A  + (size_t)row0 * K, K, M - row0, K);
        tdm_load_tile_f16(lbase + 17408u, Wt + (size_t)col0 * K, K, N - col0, K);
    }
#endif

    for (int it = 0; it < nt; ++it) {
        const int buf = it & 1;
#if USE_TDM
        if (wave == 0) {
            if (it + 1 < nt) {
                const int k0 = (it + 1) * 32;
                tdm_load_tile_f16(lbase + (buf ? 0u : 8704u),
                                  A + (size_t)row0 * K + k0, K - k0, M - row0, K);
                tdm_load_tile_f16(lbase + (buf ? 17408u : 26112u),
                                  Wt + (size_t)col0 * K + k0, K - k0, N - col0, K);
                __builtin_amdgcn_s_wait_tensorcnt(2);   // current tile's 2 DMAs done
            } else {
                __builtin_amdgcn_s_wait_tensorcnt(0);
            }
        }
        __syncthreads();
        const _Float16* As = lds + (buf ? 4352 : 0);
        const _Float16* Bs = lds + (buf ? 13056 : 8704);
#else
        const int k0 = it * 32;
        // VALU staging fallback (single buffer)
        for (int idx = tid; idx < 128 * 32; idx += 256) {
            int r = idx >> 5, kk = idx & 31;
            int gm = row0 + r, gn = col0 + r;
            lds[r * LPAD + kk] =
                (gm < M) ? A[(size_t)gm * K + k0 + kk] : (_Float16)0.0f;
            lds[4352 + r * LPAD + kk] =
                (gn < N) ? Wt[(size_t)gn * K + k0 + kk] : (_Float16)0.0f;
        }
        __syncthreads();
        const _Float16* As = lds;
        const _Float16* Bs = lds + 4352;
#endif
        // fragment gather per the 16-bit 16x32 A-operand layout (ISA 7.12.2)
        v16h afrag[2], bfrag[4];
        #pragma unroll
        for (int mi = 0; mi < 2; ++mi) {
            int m = wr * 32 + mi * 16 + lane_lo;
            #pragma unroll
            for (int p = 0; p < 8; ++p) {
                int kk = (p < 4) ? (hi * 8 + 2 * p) : (16 + hi * 8 + 2 * (p - 4));
                afrag[mi][2 * p]     = As[m * LPAD + kk];
                afrag[mi][2 * p + 1] = As[m * LPAD + kk + 1];
            }
        }
        #pragma unroll
        for (int ni = 0; ni < 4; ++ni) {
            int n = wc * 64 + ni * 16 + lane_lo;
            #pragma unroll
            for (int p = 0; p < 8; ++p) {
                int kk = (p < 4) ? (hi * 8 + 2 * p) : (16 + hi * 8 + 2 * (p - 4));
                bfrag[ni][2 * p]     = Bs[n * LPAD + kk];
                bfrag[ni][2 * p + 1] = Bs[n * LPAD + kk + 1];
            }
        }
        #pragma unroll
        for (int mi = 0; mi < 2; ++mi)
            #pragma unroll
            for (int ni = 0; ni < 4; ++ni)
                acc[mi][ni] = __builtin_amdgcn_wmma_f32_16x16x32_f16(
                    false, afrag[mi], false, bfrag[ni],
                    (short)0, acc[mi][ni], false, false);
        __syncthreads();
    }

    // epilogue: D layout — VGPR r: M = r + (lane>=16 ? 8 : 0), N = lane&15
    #pragma unroll
    for (int mi = 0; mi < 2; ++mi) {
        #pragma unroll
        for (int ni = 0; ni < 4; ++ni) {
            int gn = col0 + wc * 64 + ni * 16 + lane_lo;
            #pragma unroll
            for (int r = 0; r < 8; ++r) {
                int gm = row0 + wr * 32 + mi * 16 + hi * 8 + r;
                if (gm < M && gn < N) {
                    float v = acc[mi][ni][r];
                    if (bias) v += bias[gn];
                    if (act == 1)
                        v = 0.5f * v * (1.0f + erff(v * 0.70710678118654752f));
                    else if (act == 2)
                        v = fmaxf(v, 0.0f);
                    if (Cf) Cf[(size_t)gm * N + gn] = v;
                    if (Ch) Ch[(size_t)gm * N + gn] = (_Float16)v;
                }
            }
        }
    }
}

// ---------------------------------------------------------------------------
// fp32 [K,N] -> f16 [N,K] transposed convert (LDS-tiled, coalesced both ways).
// grid (N/32, K/32, batch), block (32,8). Batch stride = K*N.
// ---------------------------------------------------------------------------
__global__ __launch_bounds__(256) void transpose_f32_to_f16(
    const float* __restrict__ src, _Float16* __restrict__ dst, int K, int N)
{
    __shared__ float tile[32][33];
    const int nb = blockIdx.x * 32, kb = blockIdx.y * 32;
    const float* s = src + (size_t)blockIdx.z * K * N;
    _Float16* d = dst + (size_t)blockIdx.z * K * N;
    for (int i = threadIdx.y; i < 32; i += 8) {
        int k = kb + i, n = nb + threadIdx.x;
        tile[i][threadIdx.x] = (k < K && n < N) ? s[(size_t)k * N + n] : 0.0f;
    }
    __syncthreads();
    for (int i = threadIdx.y; i < 32; i += 8) {
        int n = nb + i, k = kb + threadIdx.x;
        if (n < N && k < K) d[(size_t)n * K + k] = (_Float16)tile[threadIdx.x][i];
    }
}

// fp32 -> f16 straight copy (not used for weights; kept for completeness)

// ---------------------------------------------------------------------------
// Sliding-window attention. Grid (S/256, NH, B), 256 threads. K/V context
// (768x64 f16 = 192 KB) staged into dynamic LDS (<= 320 KB WGP LDS).
// One query row per thread, two-pass softmax, fp32 accumulation.
// q,k,v,out are f16 [B*S, 768] with heads interleaved.
// ---------------------------------------------------------------------------
__global__ __launch_bounds__(256) void sliding_attn_kernel(
    const _Float16* __restrict__ q, const _Float16* __restrict__ k,
    const _Float16* __restrict__ v, const int* __restrict__ amask,
    _Float16* __restrict__ out, int S)
{
    extern __shared__ char smem[];
    _Float16* Ks = (_Float16*)smem;          // [768][64]
    _Float16* Vs = Ks + 768 * 64;            // [768][64]
    int* validj  = (int*)(Vs + 768 * 64);    // [768]

    const int c = WIN;
    const int n = blockIdx.x, h = blockIdx.y, b = blockIdx.z;
    const int base = n * c;
    const int tid = threadIdx.x;

    for (int idx = tid; idx < 768 * 64; idx += 256) {
        int j = idx >> 6, d = idx & 63;
        int t = base - c + j;
        _Float16 kv = (_Float16)0.0f, vv = (_Float16)0.0f;
        if (t >= 0 && t < S) {
            size_t o = ((size_t)(b * S + t)) * HID + h * HD + d;
            kv = k[o]; vv = v[o];
        }
        Ks[idx] = kv;
        Vs[idx] = vv;
    }
    for (int j = tid; j < 768; j += 256) {
        int t = base - c + j;
        validj[j] = (t >= 0 && t < S) ? amask[b * S + t] : 0;
    }
    __syncthreads();

    const int i = tid;
    const int s = base + i;
    float qr[64];
    {
        size_t qo = ((size_t)(b * S + s)) * HID + h * HD;
        #pragma unroll
        for (int d = 0; d < 64; ++d) qr[d] = (float)q[qo + d] * 0.125f;
    }
    const int jlo = i, jhi = i + 2 * c;

    float m = -1e30f;
    for (int j = jlo; j <= jhi; ++j) {
        if (!validj[j]) continue;
        float dot = 0.0f;
        #pragma unroll
        for (int d = 0; d < 64; ++d) dot += qr[d] * (float)Ks[j * 64 + d];
        m = fmaxf(m, dot);
    }
    if (m < -1e29f) m = 0.0f;

    float ssum = 0.0f;
    float o[64];
    #pragma unroll
    for (int d = 0; d < 64; ++d) o[d] = 0.0f;
    for (int j = jlo; j <= jhi; ++j) {
        if (!validj[j]) continue;
        float dot = 0.0f;
        #pragma unroll
        for (int d = 0; d < 64; ++d) dot += qr[d] * (float)Ks[j * 64 + d];
        float e = __expf(dot - m);
        ssum += e;
        #pragma unroll
        for (int d = 0; d < 64; ++d) o[d] += e * (float)Vs[j * 64 + d];
    }
    const float inv = 1.0f / ssum;
    size_t oo = ((size_t)(b * S + s)) * HID + h * HD;
    #pragma unroll
    for (int d = 0; d < 64; ++d) out[oo + d] = (_Float16)(o[d] * inv);
}

// ---------------------------------------------------------------------------
// LayerNorm over 768 with optional residual: out = LN(x + res)*g + b.
// Writes fp32 `outf` and optional f16 `outh` (next GEMM's A operand).
// ---------------------------------------------------------------------------
__global__ __launch_bounds__(256) void ln_kernel(
    const float* __restrict__ x, const float* res,
    const float* __restrict__ g, const float* __restrict__ bb,
    float* outf, _Float16* outh, int T)
{
    __shared__ float red[256];
    const int t = blockIdx.x;
    const int tid = threadIdx.x;
    float vals[3];
    float s = 0.0f;
    #pragma unroll
    for (int i = 0; i < 3; ++i) {
        int cidx = tid + i * 256;
        float vv = x[(size_t)t * HID + cidx];
        if (res) vv += res[(size_t)t * HID + cidx];
        vals[i] = vv; s += vv;
    }
    red[tid] = s; __syncthreads();
    for (int off = 128; off > 0; off >>= 1) {
        if (tid < off) red[tid] += red[tid + off];
        __syncthreads();
    }
    float mu = red[0] * (1.0f / HID);
    __syncthreads();
    float s2 = 0.0f;
    #pragma unroll
    for (int i = 0; i < 3; ++i) { float d = vals[i] - mu; s2 += d * d; }
    red[tid] = s2; __syncthreads();
    for (int off = 128; off > 0; off >>= 1) {
        if (tid < off) red[tid] += red[tid + off];
        __syncthreads();
    }
    float rstd = rsqrtf(red[0] * (1.0f / HID) + 1e-5f);
    #pragma unroll
    for (int i = 0; i < 3; ++i) {
        int cidx = tid + i * 256;
        float vo = (vals[i] - mu) * rstd * g[cidx] + bb[cidx];
        outf[(size_t)t * HID + cidx] = vo;
        if (outh) outh[(size_t)t * HID + cidx] = (_Float16)vo;
    }
}

// embeddings: out[t,c] = word_emb[ids[t],c] + pos_emb[s+2,c] + tt_emb[c]
__global__ __launch_bounds__(256) void embed_kernel(
    const int* __restrict__ ids, const float* __restrict__ we,
    const float* __restrict__ pe, const float* __restrict__ tte,
    float* __restrict__ out, int S)
{
    const int t = blockIdx.x;
    const int s = t % S;
    const int id = ids[t];
    const int tid = threadIdx.x;
    #pragma unroll
    for (int i = 0; i < 3; ++i) {
        int cidx = tid + i * 256;
        out[(size_t)t * HID + cidx] =
            we[(size_t)id * HID + cidx] + pe[(size_t)(s + 2) * HID + cidx] + tte[cidx];
    }
}

// span pooling -> f16 embedding rows (A operand of the projection GEMM)
__global__ __launch_bounds__(256) void span_pool_kernel(
    const float* __restrict__ seq, const float* __restrict__ masks,
    const int* __restrict__ sidx, int S, _Float16* __restrict__ emb)
{
    __shared__ float red[256];
    const int t = blockIdx.x;       // 0..15
    const int tid = threadIdx.x;
    const int bsel = sidx[t];
    float ls = 0.0f;
    for (int s = tid; s < S; s += 256) ls += masks[(size_t)t * S + s];
    red[tid] = ls; __syncthreads();
    for (int off = 128; off > 0; off >>= 1) {
        if (tid < off) red[tid] += red[tid + off];
        __syncthreads();
    }
    float len = fmaxf(red[0], 1e-9f);
    float acc0 = 0.0f, acc1 = 0.0f, acc2 = 0.0f;
    for (int s = 0; s < S; ++s) {
        float w = masks[(size_t)t * S + s];
        size_t o = ((size_t)(bsel * S + s)) * HID + tid;
        acc0 += w * seq[o];
        acc1 += w * seq[o + 256];
        acc2 += w * seq[o + 512];
    }
    emb[(size_t)t * HID + tid]       = (_Float16)(acc0 / len);
    emb[(size_t)t * HID + tid + 256] = (_Float16)(acc1 / len);
    emb[(size_t)t * HID + tid + 512] = (_Float16)(acc2 / len);
}

// row L2 normalize over 128; grid 16, block 128
__global__ __launch_bounds__(128) void l2norm_kernel(
    const float* __restrict__ in, float* __restrict__ out)
{
    __shared__ float red[128];
    const int t = blockIdx.x;
    const int tid = threadIdx.x;
    float vv = in[t * 128 + tid];
    red[tid] = vv * vv; __syncthreads();
    for (int off = 64; off > 0; off >>= 1) {
        if (tid < off) red[tid] += red[tid + off];
        __syncthreads();
    }
    float nrm = fmaxf(sqrtf(red[0]), 1e-12f);
    out[t * 128 + tid] = vv / nrm;
}

// ---------------------------------------------------------------------------
extern "C" void kernel_launch(void* const* d_in, const int* in_sizes, int n_in,
                              void* d_out, int out_size, void* d_ws, size_t ws_size,
                              hipStream_t stream)
{
    (void)in_sizes; (void)n_in; (void)out_size; (void)ws_size;
    const int*   doc_ids   = (const int*)  d_in[0];
    const int*   doc_mask  = (const int*)  d_in[1];
    const int*   sum_ids   = (const int*)  d_in[2];
    const int*   sum_mask  = (const int*)  d_in[3];
    const float* og_masks  = (const float*)d_in[4];
    const float* llm_masks = (const float*)d_in[5];
    const int*   sidx      = (const int*)  d_in[6];
    const float* we        = (const float*)d_in[7];
    const float* pe        = (const float*)d_in[8];
    const float* tte       = (const float*)d_in[9];
    const float* ln_emb_g  = (const float*)d_in[10];
    const float* ln_emb_b  = (const float*)d_in[11];
    const float* Wq        = (const float*)d_in[12];
    const float* bq        = (const float*)d_in[13];
    const float* Wk        = (const float*)d_in[14];
    const float* bk        = (const float*)d_in[15];
    const float* Wv        = (const float*)d_in[16];
    const float* bv        = (const float*)d_in[17];
    const float* Wo        = (const float*)d_in[18];
    const float* bo        = (const float*)d_in[19];
    const float* ln1_g     = (const float*)d_in[20];
    const float* ln1_b     = (const float*)d_in[21];
    const float* W1        = (const float*)d_in[22];
    const float* b1        = (const float*)d_in[23];
    const float* W2        = (const float*)d_in[24];
    const float* b2        = (const float*)d_in[25];
    const float* ln2_g     = (const float*)d_in[26];
    const float* ln2_b     = (const float*)d_in[27];
    const float* pW1       = (const float*)d_in[28];
    const float* pb1       = (const float*)d_in[29];
    const float* pW2       = (const float*)d_in[30];
    const float* pb2       = (const float*)d_in[31];
    float* out = (float*)d_out;

    // ---- workspace layout (bump allocator, 256B aligned) ----
    const size_t TD = 2 * 2048, TS = 2 * 512;
    const size_t H2 = (size_t)HID * HID;          // 589824
    const size_t HF = (size_t)HID * FF;           // 2359296
    char* p = (char*)d_ws;
    auto alloc = [&](size_t bytes) -> char* {
        char* r = p; p += (bytes + 255) & ~(size_t)255; return r;
    };
    _Float16* wtq  = (_Float16*)alloc(12 * H2 * 2);
    _Float16* wtk  = (_Float16*)alloc(12 * H2 * 2);
    _Float16* wtv  = (_Float16*)alloc(12 * H2 * 2);
    _Float16* wto  = (_Float16*)alloc(12 * H2 * 2);
    _Float16* wt1  = (_Float16*)alloc(12 * HF * 2);   // [3072,768] per layer
    _Float16* wt2  = (_Float16*)alloc(12 * HF * 2);   // [768,3072] per layer
    _Float16* wtp1 = (_Float16*)alloc(H2 * 2);
    _Float16* wtp2 = (_Float16*)alloc((size_t)HID * PDIM * 2);
    float*    xdoc = (float*)   alloc(TD * HID * 4);
    float*    xsum = (float*)   alloc(TS * HID * 4);
    float*    tmp  = (float*)   alloc(TD * HID * 4);
    _Float16* x16  = (_Float16*)alloc(TD * HID * 2);
    _Float16* q16  = (_Float16*)alloc(TD * HID * 2);
    _Float16* k16  = (_Float16*)alloc(TD * HID * 2);
    _Float16* v16  = (_Float16*)alloc(TD * HID * 2);
    _Float16* a16  = (_Float16*)alloc(TD * HID * 2);
    _Float16* h16  = (_Float16*)alloc(TD * FF * 2);
    _Float16* embh = (_Float16*)alloc(16 * HID * 2);
    _Float16* p1h  = (_Float16*)alloc(16 * HID * 2);
    float*    proj2= (float*)   alloc(16 * PDIM * 4);

    // ---- weight convert + transpose to f16 [N,K] ----
    {
        dim3 tb(32, 8);
        transpose_f32_to_f16<<<dim3(HID/32, HID/32, 12), tb, 0, stream>>>(Wq, wtq, HID, HID);
        transpose_f32_to_f16<<<dim3(HID/32, HID/32, 12), tb, 0, stream>>>(Wk, wtk, HID, HID);
        transpose_f32_to_f16<<<dim3(HID/32, HID/32, 12), tb, 0, stream>>>(Wv, wtv, HID, HID);
        transpose_f32_to_f16<<<dim3(HID/32, HID/32, 12), tb, 0, stream>>>(Wo, wto, HID, HID);
        transpose_f32_to_f16<<<dim3(FF/32,  HID/32, 12), tb, 0, stream>>>(W1, wt1, HID, FF);
        transpose_f32_to_f16<<<dim3(HID/32, FF/32,  12), tb, 0, stream>>>(W2, wt2, FF, HID);
        transpose_f32_to_f16<<<dim3(HID/32, HID/32, 1),  tb, 0, stream>>>(pW1, wtp1, HID, HID);
        transpose_f32_to_f16<<<dim3(PDIM/32, HID/32, 1), tb, 0, stream>>>(pW2, wtp2, HID, PDIM);
    }

    const size_t gemm_lds = 34816;   // 2x(A 8704B) + 2x(B 8704B)
    auto gemm = [&](const _Float16* A, const _Float16* Wt, const float* bias,
                    float* Cf, _Float16* Ch, int M, int N, int K, int act) {
        dim3 grid((N + 127) / 128, (M + 127) / 128);
        gemm_tdm_wmma<<<grid, 256, gemm_lds, stream>>>(A, Wt, bias, Cf, Ch, M, N, K, act);
    };

    const size_t attn_lds = (size_t)768 * 64 * 2 * sizeof(_Float16) + 768 * sizeof(int);

    auto encode = [&](const int* ids, const int* amask, int B, int S, float* x) {
        int T = B * S;
        embed_kernel<<<T, 256, 0, stream>>>(ids, we, pe, tte, tmp, S);
        ln_kernel<<<T, 256, 0, stream>>>(tmp, nullptr, ln_emb_g, ln_emb_b, x, x16, T);
        for (int l = 0; l < 12; ++l) {
            const size_t o2 = (size_t)l * H2;
            const size_t oF = (size_t)l * HF;
            gemm(x16, wtq + o2, bq + l * HID, nullptr, q16, T, HID, HID, 0);
            gemm(x16, wtk + o2, bk + l * HID, nullptr, k16, T, HID, HID, 0);
            gemm(x16, wtv + o2, bv + l * HID, nullptr, v16, T, HID, HID, 0);
            dim3 ag(S / WIN, NH, B);
            sliding_attn_kernel<<<ag, 256, attn_lds, stream>>>(q16, k16, v16, amask, a16, S);
            gemm(a16, wto + o2, bo + l * HID, tmp, nullptr, T, HID, HID, 0);
            ln_kernel<<<T, 256, 0, stream>>>(tmp, x, ln1_g + l * HID, ln1_b + l * HID, x, x16, T);
            gemm(x16, wt1 + oF, b1 + l * FF, nullptr, h16, T, FF, HID, 1);   // GELU fused
            gemm(h16, wt2 + oF, b2 + l * HID, tmp, nullptr, T, HID, FF, 0);
            ln_kernel<<<T, 256, 0, stream>>>(tmp, x, ln2_g + l * HID, ln2_b + l * HID, x, x16, T);
        }
    };

    encode(doc_ids, doc_mask, 2, 2048, xdoc);
    encode(sum_ids, sum_mask, 2, 512, xsum);

    // human spans from doc encoding
    span_pool_kernel<<<16, 256, 0, stream>>>(xdoc, og_masks, sidx, 2048, embh);
    gemm(embh, wtp1, pb1, nullptr, p1h, 16, HID, HID, 2);      // ReLU fused
    gemm(p1h, wtp2, pb2, proj2, nullptr, 16, PDIM, HID, 0);
    l2norm_kernel<<<16, 128, 0, stream>>>(proj2, out);

    // llm spans from summary encoding
    span_pool_kernel<<<16, 256, 0, stream>>>(xsum, llm_masks, sidx, 512, embh);
    gemm(embh, wtp1, pb1, nullptr, p1h, 16, HID, HID, 2);
    gemm(p1h, wtp2, pb2, proj2, nullptr, 16, PDIM, HID, 0);
    l2norm_kernel<<<16, 128, 0, stream>>>(proj2, out + 16 * PDIM);
}
